// RecurrentClassifier_88622355185847
// MI455X (gfx1250) — matrix operand order repaired and checked by reading, not compiled, not run on hardware
//
#include <hip/hip_runtime.h>

typedef __attribute__((ext_vector_type(16))) _Float16 v16h;
typedef __attribute__((ext_vector_type(8)))  _Float16 v8h;
typedef __attribute__((ext_vector_type(8)))  float    v8f;

// Problem sizes
constexpr int ENC = 512;
constexpr int L1  = 1024;
constexpr int NC  = 10;
constexpr int T   = 100;
constexpr int B   = 16384;
constexpr float BETA   = 0.9f;
constexpr float THRESH = 1.0f;

// -------------------------------------------------------------------------
// Utility kernels
// -------------------------------------------------------------------------
__global__ void zero_f32_kernel(float* __restrict__ p, int n) {
    for (int i = blockIdx.x * blockDim.x + threadIdx.x; i < n;
         i += gridDim.x * blockDim.x)
        p[i] = 0.0f;
}

// dst[k*jcount + j] = (f16) src[j*kcount + k]   (k-major transposed copy)
__global__ void transpose_f32_to_f16_kernel(const float* __restrict__ src,
                                            _Float16* __restrict__ dst,
                                            int jcount, int kcount) {
    int idx = blockIdx.x * 256 + threadIdx.x;
    if (idx < jcount * kcount) {
        int k = idx / jcount;
        int j = idx - k * jcount;
        dst[idx] = (_Float16)src[(size_t)j * kcount + k];
    }
}

// -------------------------------------------------------------------------
// One-time GEMM: cur1 = x @ fc1_w^T + fc1_b    (M=16384, N=1024, K=512)
// A = x (f32, converted to f16 while staging), B = fc1T (f16, k-major)
// Workgroup: 256 threads (8 waves), 128x128 tile, K-step 32.
// Wave tiling: 4 waves in M x 2 waves in N; per wave 32M x 64N
// (2 A-frags x 4 B-frags -> 8 WMMAs, every fragment reused in-regs).
// -------------------------------------------------------------------------
__global__ __launch_bounds__(256) void gemm_fc1_kernel(
    const float* __restrict__ x,       // [B][ENC] f32
    const _Float16* __restrict__ fc1T, // [ENC][L1] f16 (k-major)
    const float* __restrict__ fc1_b,   // [L1]
    float* __restrict__ cur1)          // [B][L1]
{
    __shared__ __attribute__((aligned(16))) _Float16 Alds[128][40];  // 32 K + pad
    __shared__ __attribute__((aligned(16))) _Float16 Blds[32][136];  // 128 N + pad

    const int tid  = threadIdx.x;
    const int lane = tid & 31;
    const int wave = tid >> 5;
    const int wm   = wave & 3;        // M sub-tile (32 rows)
    const int wn   = wave >> 2;       // N sub-tile (64 cols)
    const int b0 = blockIdx.x * 128;
    const int n0 = blockIdx.y * 128;

    const v8f vzero = {0.f,0.f,0.f,0.f,0.f,0.f,0.f,0.f};
    v8f acc[2][4];
#pragma unroll
    for (int mi = 0; mi < 2; mi++)
#pragma unroll
        for (int ni = 0; ni < 4; ni++) acc[mi][ni] = vzero;

    const int ar = tid >> 1;           // 0..127 (A row)
    const int ac = (tid & 1) * 16;     // 0/16   (A col chunk)
    const int br = tid >> 3;           // 0..31  (B k row)
    const int bc = (tid & 7) * 16;     // 0..112 (B n chunk)

    const float*    agp = x    + (size_t)(b0 + ar) * ENC + ac;
    const _Float16* bgp = fc1T + (size_t)br * L1 + n0 + bc;

    // Preload K-tile 0 into registers
    v8h c0, c1;
#pragma unroll
    for (int i = 0; i < 8; i++) {
        c0[i] = (_Float16)agp[i];
        c1[i] = (_Float16)agp[8 + i];
    }
    uint4 bv0 = *(const uint4*)(bgp);
    uint4 bv1 = *(const uint4*)(bgp + 8);

    for (int k0 = 0; k0 < ENC; k0 += 32) {
        __syncthreads();
        *(v8h*)&Alds[ar][ac]       = c0;
        *(v8h*)&Alds[ar][ac + 8]   = c1;
        *(uint4*)&Blds[br][bc]     = bv0;
        *(uint4*)&Blds[br][bc + 8] = bv1;
        __syncthreads();

        // Prefetch next K-tile (wraps on last iteration; in-bounds)
        const int kn = (k0 + 32) & (ENC - 1);
#pragma unroll
        for (int i = 0; i < 8; i++) {
            c0[i] = (_Float16)agp[kn + i];
            c1[i] = (_Float16)agp[kn + 8 + i];
        }
        bv0 = *(const uint4*)(bgp + (size_t)kn * L1);
        bv1 = *(const uint4*)(bgp + (size_t)kn * L1 + 8);

        // Fragments
        const int koff = (lane >> 4) * 8;
        v16h afrag[2];
#pragma unroll
        for (int mi = 0; mi < 2; mi++) {
            const _Float16* ap = &Alds[wm * 32 + mi * 16 + (lane & 15)][koff];
            v8h alo = *(const v8h*)ap;
            v8h ahi = *(const v8h*)(ap + 16);
            afrag[mi] = __builtin_shufflevector(alo, ahi,
                            0,1,2,3,4,5,6,7,8,9,10,11,12,13,14,15);
        }
        v16h bfrag[4];
#pragma unroll
        for (int ni = 0; ni < 4; ni++) {
            const int bn = wn * 64 + ni * 16 + (lane & 15);
#pragma unroll
            for (int s = 0; s < 8; s++) {
                bfrag[ni][s]     = Blds[koff + s][bn];
                bfrag[ni][8 + s] = Blds[16 + koff + s][bn];
            }
        }
#pragma unroll
        for (int mi = 0; mi < 2; mi++)
#pragma unroll
            for (int ni = 0; ni < 4; ni++)
                acc[mi][ni] = __builtin_amdgcn_wmma_f32_16x16x32_f16(
                    false, afrag[mi], false, bfrag[ni], (short)0,
                    acc[mi][ni], false, false);
    }

    const int rowhalf = 8 * (lane >> 4);
    const int colL    = lane & 15;
#pragma unroll
    for (int mi = 0; mi < 2; mi++)
#pragma unroll
        for (int ni = 0; ni < 4; ni++)
#pragma unroll
            for (int r = 0; r < 8; r++) {
                int brow = b0 + wm * 32 + mi * 16 + rowhalf + r;
                int j    = n0 + wn * 64 + ni * 16 + colL;
                cur1[(size_t)brow * L1 + j] = acc[mi][ni][r] + fc1_b[j];
            }
}

// -------------------------------------------------------------------------
// Per-step layer-1 kernel: WMMA GEMM acc = spk_prev @ v1_w^T, fused LIF:
// mem1 = BETA*mem1 + cur1 + acc + v1_b - (mem1_old>1)
// spk_next = heaviside(mem1 - 1)      (M=16384, N=1024, K=1024)
// -------------------------------------------------------------------------
__global__ __launch_bounds__(256) void step1_kernel(
    const _Float16* __restrict__ spk_prev, // [B][L1] f16
    const _Float16* __restrict__ wT,       // [L1][L1] f16 (k-major: wT[k][j])
    const float* __restrict__ cur1,        // [B][L1]
    float* __restrict__ mem1,              // [B][L1] (in/out)
    const float* __restrict__ v1_b,        // [L1]
    _Float16* __restrict__ spk_next)       // [B][L1] f16
{
    __shared__ __attribute__((aligned(16))) _Float16 Alds[128][40];
    __shared__ __attribute__((aligned(16))) _Float16 Blds[32][136];

    const int tid  = threadIdx.x;
    const int lane = tid & 31;
    const int wave = tid >> 5;
    const int wm   = wave & 3;
    const int wn   = wave >> 2;
    const int b0 = blockIdx.x * 128;
    const int n0 = blockIdx.y * 128;

    const v8f vzero = {0.f,0.f,0.f,0.f,0.f,0.f,0.f,0.f};
    v8f acc[2][4];
#pragma unroll
    for (int mi = 0; mi < 2; mi++)
#pragma unroll
        for (int ni = 0; ni < 4; ni++) acc[mi][ni] = vzero;

    const int ar = tid >> 1;
    const int ac = (tid & 1) * 16;
    const int br = tid >> 3;
    const int bc = (tid & 7) * 16;

    const _Float16* agp = spk_prev + (size_t)(b0 + ar) * L1 + ac;
    const _Float16* bgp = wT + (size_t)br * L1 + n0 + bc;

    // Preload K-tile 0
    uint4 av0 = *(const uint4*)(agp);
    uint4 av1 = *(const uint4*)(agp + 8);
    uint4 bv0 = *(const uint4*)(bgp);
    uint4 bv1 = *(const uint4*)(bgp + 8);

    for (int k0 = 0; k0 < L1; k0 += 32) {
        __syncthreads();
        *(uint4*)&Alds[ar][ac]     = av0;
        *(uint4*)&Alds[ar][ac + 8] = av1;
        *(uint4*)&Blds[br][bc]     = bv0;
        *(uint4*)&Blds[br][bc + 8] = bv1;
        __syncthreads();

        // Prefetch next K-tile (wraps on last iteration; stays in-bounds)
        const int kn = (k0 + 32) & (L1 - 1);
        av0 = *(const uint4*)(agp + kn);
        av1 = *(const uint4*)(agp + kn + 8);
        bv0 = *(const uint4*)(bgp + (size_t)kn * L1);
        bv1 = *(const uint4*)(bgp + (size_t)kn * L1 + 8);

        // Fragments (each reused 4x / 2x across the 8 WMMAs)
        const int koff = (lane >> 4) * 8;
        v16h afrag[2];
#pragma unroll
        for (int mi = 0; mi < 2; mi++) {
            const _Float16* ap = &Alds[wm * 32 + mi * 16 + (lane & 15)][koff];
            v8h alo = *(const v8h*)ap;
            v8h ahi = *(const v8h*)(ap + 16);
            afrag[mi] = __builtin_shufflevector(alo, ahi,
                            0,1,2,3,4,5,6,7,8,9,10,11,12,13,14,15);
        }
        v16h bfrag[4];
#pragma unroll
        for (int ni = 0; ni < 4; ni++) {
            const int bn = wn * 64 + ni * 16 + (lane & 15);
#pragma unroll
            for (int s = 0; s < 8; s++) {
                bfrag[ni][s]     = Blds[koff + s][bn];
                bfrag[ni][8 + s] = Blds[16 + koff + s][bn];
            }
        }
#pragma unroll
        for (int mi = 0; mi < 2; mi++)
#pragma unroll
            for (int ni = 0; ni < 4; ni++)
                acc[mi][ni] = __builtin_amdgcn_wmma_f32_16x16x32_f16(
                    false, afrag[mi], false, bfrag[ni], (short)0,
                    acc[mi][ni], false, false);
    }

    // Fused LIF epilogue. D layout: reg r -> M = r + 8*(lane>=16), N = lane&15
    const int rowhalf = 8 * (lane >> 4);
    const int colL    = lane & 15;
#pragma unroll
    for (int mi = 0; mi < 2; mi++)
#pragma unroll
        for (int ni = 0; ni < 4; ni++)
#pragma unroll
            for (int r = 0; r < 8; r++) {
                int brow = b0 + wm * 32 + mi * 16 + rowhalf + r;
                int j    = n0 + wn * 64 + ni * 16 + colL;
                size_t idx = (size_t)brow * L1 + j;
                float mold  = mem1[idx];
                float reset = (mold > THRESH) ? THRESH : 0.0f;
                float mnew  = BETA * mold + cur1[idx] + acc[mi][ni][r]
                              + v1_b[j] - reset;
                mem1[idx] = mnew;
                spk_next[idx] = (_Float16)((mnew - THRESH > 0.0f) ? 1.0f : 0.0f);
            }
}

// -------------------------------------------------------------------------
// Per-step layer-2 kernel: cur2 = spk1 @ fc2_w^T + fc2_b (N=10, K=1024),
// recurrent LIF on [B][NC]; state lives directly in d_out (spk_rec/mem_rec).
// One wave per batch row.
// -------------------------------------------------------------------------
__global__ __launch_bounds__(256) void step2_kernel(
    const _Float16* __restrict__ spk1,  // [B][L1] f16 (this step's spikes)
    const float* __restrict__ fc2_w,    // [NC][L1]
    const float* __restrict__ fc2_b,    // [NC]
    const float* __restrict__ v2_w,     // [NC][NC]
    const float* __restrict__ v2_b,     // [NC]
    float* __restrict__ out,            // spk_rec | mem_rec, each [T][B][NC]
    int t)
{
    const int lane = threadIdx.x & 31;
    const int wave = threadIdx.x >> 5;
    const int b = blockIdx.x * 8 + wave;

    const _Float16* row = spk1 + (size_t)b * L1 + lane * 32;
    v8h h0 = *(const v8h*)(row);
    v8h h1 = *(const v8h*)(row + 8);
    v8h h2 = *(const v8h*)(row + 16);
    v8h h3 = *(const v8h*)(row + 24);

    const float* wbase = fc2_w + lane * 32;
    float s[NC];
#pragma unroll
    for (int n = 0; n < NC; n++) {
        const float* wr = wbase + n * L1;
        float sn = 0.0f;
#pragma unroll
        for (int i = 0; i < 8; i++) {
            sn += (float)h0[i] * wr[i];
            sn += (float)h1[i] * wr[8 + i];
            sn += (float)h2[i] * wr[16 + i];
            sn += (float)h3[i] * wr[24 + i];
        }
#pragma unroll
        for (int off = 16; off > 0; off >>= 1)
            sn += __shfl_xor(sn, off, 32);
        s[n] = sn;
    }

    if (lane < NC) {
        const int n = lane;
        float cur2 = 0.0f;
#pragma unroll
        for (int i = 0; i < NC; i++) cur2 = (n == i) ? s[i] : cur2;
        cur2 += fc2_b[n];

        const size_t memrec_off = (size_t)T * B * NC;
        float m2old = 0.0f, rec = 0.0f;
        if (t > 0) {
            const float* sprev = out + (size_t)(t - 1) * B * NC + (size_t)b * NC;
            const float* mprev = out + memrec_off + (size_t)(t - 1) * B * NC
                                 + (size_t)b * NC;
            m2old = mprev[n];
#pragma unroll
            for (int m = 0; m < NC; m++) rec += sprev[m] * v2_w[n * NC + m];
        }
        float reset = (m2old > THRESH) ? THRESH : 0.0f;
        float m2 = BETA * m2old + cur2 + rec + v2_b[n] - reset;
        out[(size_t)t * B * NC + (size_t)b * NC + n] =
            (m2 - THRESH > 0.0f) ? 1.0f : 0.0f;
        out[memrec_off + (size_t)t * B * NC + (size_t)b * NC + n] = m2;
    }
}

// -------------------------------------------------------------------------
// Host launcher
// -------------------------------------------------------------------------
extern "C" void kernel_launch(void* const* d_in, const int* in_sizes, int n_in,
                              void* d_out, int out_size, void* d_ws, size_t ws_size,
                              hipStream_t stream) {
    (void)in_sizes; (void)n_in; (void)out_size; (void)ws_size;

    const float* x     = (const float*)d_in[0];
    const float* fc1_w = (const float*)d_in[1];
    const float* fc1_b = (const float*)d_in[2];
    const float* v1_w  = (const float*)d_in[3];
    const float* v1_b  = (const float*)d_in[4];
    const float* fc2_w = (const float*)d_in[5];
    const float* fc2_b = (const float*)d_in[6];
    const float* v2_w  = (const float*)d_in[7];
    const float* v2_b  = (const float*)d_in[8];
    float* out = (float*)d_out;

    // Workspace layout (bytes)
    char* ws = (char*)d_ws;
    constexpr size_t SPK_BYTES  = (size_t)B * L1 * 2;   // 32 MB (f16)
    constexpr size_t F32_BYTES  = (size_t)B * L1 * 4;   // 64 MB
    _Float16* spkA = (_Float16*)(ws);
    _Float16* spkB = (_Float16*)(ws + SPK_BYTES);
    float*    cur1 = (float*)   (ws + 2 * SPK_BYTES);
    float*    mem1 = (float*)   (ws + 2 * SPK_BYTES + F32_BYTES);
    _Float16* wT   = (_Float16*)(ws + 2 * SPK_BYTES + 2 * F32_BYTES);
    _Float16* fc1T = (_Float16*)(ws + 2 * SPK_BYTES + 2 * F32_BYTES
                                 + (size_t)L1 * L1 * 2);

    // Prep: zero state, transpose weights to k-major f16
    zero_f32_kernel<<<4096, 256, 0, stream>>>(mem1, B * L1);
    zero_f32_kernel<<<2048, 256, 0, stream>>>((float*)spkA, B * L1 / 2);
    transpose_f32_to_f16_kernel<<<(L1 * L1 + 255) / 256, 256, 0, stream>>>(
        v1_w, wT, L1, L1);
    transpose_f32_to_f16_kernel<<<(ENC * L1 + 255) / 256, 256, 0, stream>>>(
        fc1_w, fc1T, L1, ENC);

    // Loop-invariant cur1 = x @ fc1_w^T + fc1_b
    dim3 gemm_grid(B / 128, L1 / 128);
    gemm_fc1_kernel<<<gemm_grid, 256, 0, stream>>>(x, fc1T, fc1_b, cur1);

    // Time loop: recurrent WMMA GEMM + fused LIF, then layer 2
    for (int t = 0; t < T; t++) {
        _Float16* prev = (t & 1) ? spkB : spkA;
        _Float16* next = (t & 1) ? spkA : spkB;
        step1_kernel<<<gemm_grid, 256, 0, stream>>>(prev, wT, cur1, mem1,
                                                    v1_b, next);
        step2_kernel<<<B / 8, 256, 0, stream>>>(next, fc2_w, fc2_b, v2_w,
                                                v2_b, out, t);
    }
}
